// Attention_45286135169661
// MI455X (gfx1250) — matrix operand order repaired and checked by reading, not compiled, not run on hardware
//
#include <hip/hip_runtime.h>
#include <hip/hip_bf16.h>

// ---------------------------------------------------------------------------
// Types / helpers
// ---------------------------------------------------------------------------
typedef __bf16 bf16_t;
typedef __attribute__((ext_vector_type(8)))  __bf16 v8bf;
typedef __attribute__((ext_vector_type(16))) __bf16 v16bf;
typedef __attribute__((ext_vector_type(8)))  float  v8f;
typedef int v4i_b __attribute__((vector_size(16)));   // int4, matches builtin proto

#define SEQL 2048
#define DIMM 4096
#define NH   32
#define NKV  8
#define HD   128
#define ATT_SCALE 0.08838834764831845f  // 1/sqrt(128)

// CDNA5 async global->LDS path (bypasses VGPRs, tracked by ASYNCcnt)
#if defined(__AMDGCN__) && __has_builtin(__builtin_amdgcn_global_load_async_to_lds_b128) && __has_builtin(__builtin_amdgcn_s_wait_asynccnt)
#define USE_ASYNC_LDS 1
#else
#define USE_ASYNC_LDS 0
#endif

__device__ __forceinline__ void async_copy_b128(const void* g, void* l) {
#if USE_ASYNC_LDS
    __builtin_amdgcn_global_load_async_to_lds_b128(
        (__attribute__((address_space(1))) v4i_b*)(uintptr_t)g,
        (__attribute__((address_space(3))) v4i_b*)(unsigned)(uintptr_t)l,
        0, 0);
#else
    *(uint4*)l = *(const uint4*)g;   // fallback: sync copy through VGPRs
#endif
}

template <int N>
__device__ __forceinline__ void async_wait() {
#if USE_ASYNC_LDS
    __builtin_amdgcn_s_wait_asynccnt(N);   // requires constant integer
#endif
}

__device__ __forceinline__ bf16_t f2bf(float f) {
    union { float f; unsigned u; } x; x.f = f;
    unsigned r = x.u + 0x7FFFu + ((x.u >> 16) & 1u);   // round-to-nearest-even
    unsigned short s = (unsigned short)(r >> 16);
    return __builtin_bit_cast(__bf16, s);
}

// A-matrix fragment (16x32 bf16): lane L -> row = L&15, half h = L>>4.
// elements 0..7  <- K = h*8 + e       (contiguous 16B)
// elements 8..15 <- K = 16 + h*8 + e' (contiguous 16B)
template <typename P>
__device__ __forceinline__ v16bf frag_a_load(P base) {
    int h = (threadIdx.x & 31) >> 4;
    v8bf lo = *(const v8bf*)(base + h * 8);
    v8bf hi = *(const v8bf*)(base + 16 + h * 8);
    return __builtin_shufflevector(lo, hi, 0,1,2,3,4,5,6,7,8,9,10,11,12,13,14,15);
}

// B-matrix fragment (32x16 bf16), operand stored [n][k] row-major:
// lane L -> n = L&15, half h = L>>4, elements 0..7 -> K = h*8 + e, 8..15 -> K = 16 + h*8 + e'
template <typename P>
__device__ __forceinline__ v16bf frag_b_load(P base) {
    int h = (threadIdx.x & 31) >> 4;
    v8bf lo = *(const v8bf*)(base + h * 8);
    v8bf hi = *(const v8bf*)(base + 16 + h * 8);
    return __builtin_shufflevector(lo, hi, 0,1,2,3,4,5,6,7,8,9,10,11,12,13,14,15);
}

__device__ __forceinline__ v8f wmma_bf16(v16bf a, v16bf b, v8f c) {
    return __builtin_amdgcn_wmma_f32_16x16x32_bf16(
        /*neg_a=*/false, a, /*neg_b=*/false, b,
        /*c_mod=*/(short)0, c, /*reuse_a=*/false, /*reuse_b=*/false);
}

// ---------------------------------------------------------------------------
// fp32 -> bf16 cast (4 elems / thread, float4 loads)
// ---------------------------------------------------------------------------
__global__ __launch_bounds__(256) void cast_bf16_kernel(
    const float* __restrict__ in, bf16_t* __restrict__ out, int n) {
    int i = (blockIdx.x * 256 + threadIdx.x) * 4;
    if (i + 3 < n) {
        float4 v = *(const float4*)(in + i);
        out[i + 0] = f2bf(v.x);
        out[i + 1] = f2bf(v.y);
        out[i + 2] = f2bf(v.z);
        out[i + 3] = f2bf(v.w);
    }
}

// ---------------------------------------------------------------------------
// GEMM: C[M][N] (f32) = A[M][K] (bf16) * B[N][K]^T (bf16)
// 128x128 block, BK=64, 256 threads = 8 waves (4 in M x 2 in N),
// each wave -> 32x64 tile = 2x4 frags of 16x16 f32.
// Double-buffered LDS stages filled with async global->LDS copies:
// per wave, per stage: 4 A + 4 B = 8 b128 async ops -> s_wait_asynccnt 8
// overlaps the *next* tile's DMA with the current tile's 16 WMMAs.
// ---------------------------------------------------------------------------
#define GBM 128
#define GBN 128
#define GBK 64
#define GLS 72   // LDS row stride in bf16 (64 + 8 pad)

__device__ __forceinline__ void gemm_stage(
    const bf16_t* __restrict__ A, const bf16_t* __restrict__ B,
    bf16_t* as, bf16_t* bs, int bm, int bn, int K, int k0, int tid) {
    for (int t = 0; t < 4; ++t) {
        int chunk = tid + t * 256;          // 0..1023
        int row   = chunk >> 3;             // 0..127
        int col   = (chunk & 7) << 3;       // 0..56
        async_copy_b128(A + (size_t)(bm + row) * K + k0 + col, as + row * GLS + col);
        async_copy_b128(B + (size_t)(bn + row) * K + k0 + col, bs + row * GLS + col);
    }
}

__global__ __launch_bounds__(256) void gemm_bf16_kernel(
    const bf16_t* __restrict__ A, const bf16_t* __restrict__ B,
    float* __restrict__ C, int M, int N, int K) {
    __shared__ bf16_t As[2][GBM * GLS];
    __shared__ bf16_t Bs[2][GBN * GLS];

    const int tid  = threadIdx.x;
    const int lane = tid & 31;
    const int wave = tid >> 5;
    const int wm   = wave & 3;   // 0..3  (M direction)
    const int wn   = wave >> 2;  // 0..1  (N direction)
    const int bm   = blockIdx.y * GBM;
    const int bn   = blockIdx.x * GBN;
    const int hh   = lane >> 4;
    const int nn   = lane & 15;

    v8f acc[2][4] = {};

    const int ktiles = K / GBK;
    gemm_stage(A, B, As[0], Bs[0], bm, bn, K, 0, tid);   // prologue: tile 0

    for (int kt = 0; kt < ktiles; ++kt) {
        const int b = kt & 1;
        if (kt + 1 < ktiles) {
            gemm_stage(A, B, As[b ^ 1], Bs[b ^ 1], bm, bn, K, (kt + 1) * GBK, tid);
            async_wait<8>();   // tile kt landed (ASYNCcnt completes in order)
        } else {
            async_wait<0>();
        }
        __syncthreads();

        const bf16_t* as = As[b];
        const bf16_t* bs = Bs[b];
        for (int s = 0; s < 2; ++s) {          // two 32-deep WMMA K steps
            v16bf af[2], bfr[4];
            for (int i = 0; i < 2; ++i)
                af[i] = frag_a_load(as + (wm * 32 + i * 16 + nn) * GLS + s * 32);
            for (int j = 0; j < 4; ++j)
                bfr[j] = frag_b_load(bs + (wn * 64 + j * 16 + nn) * GLS + s * 32);
            for (int i = 0; i < 2; ++i)
                for (int j = 0; j < 4; ++j)
                    acc[i][j] = wmma_bf16(af[i], bfr[j], acc[i][j]);
        }
        __syncthreads();   // compute done before buffer b is recycled
    }

    // C/D layout: VGPR r, lane L -> row = r + 8*(L>>4), col = L&15
    for (int i = 0; i < 2; ++i)
        for (int j = 0; j < 4; ++j)
            for (int r = 0; r < 8; ++r) {
                int row = bm + wm * 32 + i * 16 + r + 8 * hh;
                int col = bn + wn * 64 + j * 16 + nn;
                C[(size_t)row * N + col] = acc[i][j][r];
            }
}

// ---------------------------------------------------------------------------
// RoPE + repack: proj [SEQ][nh*128] f32 -> out [nh][SEQ][128] bf16
// ---------------------------------------------------------------------------
__global__ __launch_bounds__(256) void rope_pack_kernel(
    const float* __restrict__ proj, const float* __restrict__ cosf,
    const float* __restrict__ sinf, bf16_t* __restrict__ out, int nheads) {
    int idx = blockIdx.x * 256 + threadIdx.x;   // over SEQ*nheads*64
    int d = idx & 63;
    int h = (idx >> 6) % nheads;
    int s = idx / (64 * nheads);
    size_t base = (size_t)s * (nheads * HD) + h * HD;
    float x1 = proj[base + 2 * d];
    float x2 = proj[base + 2 * d + 1];
    float c  = cosf[s * 64 + d];
    float sn = sinf[s * 64 + d];
    size_t ob = ((size_t)h * SEQL + s) * HD;
    out[ob + 2 * d]     = f2bf(x1 * c - x2 * sn);
    out[ob + 2 * d + 1] = f2bf(x1 * sn + x2 * c);
}

// V [SEQ][NKV*128] f32 -> Vt [NKV][128][SEQ] bf16 (key-contiguous for P*V B-frags)
__global__ __launch_bounds__(256) void pack_vt_kernel(
    const float* __restrict__ v, bf16_t* __restrict__ vt) {
    int idx = blockIdx.x * 256 + threadIdx.x;   // over SEQ*NKV*HD
    int s    = idx & (SEQL - 1);
    int rest = idx >> 11;
    int d    = rest & (HD - 1);
    int h    = rest >> 7;
    vt[((size_t)h * HD + d) * SEQL + s] = f2bf(v[(size_t)s * (NKV * HD) + h * HD + d]);
}

// ---------------------------------------------------------------------------
// Flash attention: one wave per (head, 16-row query tile), KV tiles of 32.
// Qh [32][SEQ][128], Kh [8][SEQ][128], Vt [8][128][SEQ] -> Out [SEQ][4096] bf16
// ---------------------------------------------------------------------------
__global__ __launch_bounds__(128) void attn_kernel(
    const bf16_t* __restrict__ Qh, const bf16_t* __restrict__ Kh,
    const bf16_t* __restrict__ Vt, bf16_t* __restrict__ Out) {
    __shared__ bf16_t Ps[4][16 * 40];   // per-wave P staging (C-layout -> A-layout)

    const int lane  = threadIdx.x & 31;
    const int wave  = threadIdx.x >> 5;
    const int head  = blockIdx.x;
    const int qtile = blockIdx.y * 4 + wave;
    const int qbase = qtile * 16;
    const int kvh   = head >> 2;        // GQA 4:1
    const int hh    = lane >> 4;
    const int nn    = lane & 15;

    // Q fragments: 16 rows x 128 dim = 4 A-frags (K chunks of 32)
    const bf16_t* qptr = Qh + ((size_t)head * SEQL + qbase + nn) * HD;
    v16bf qf[4];
    for (int c = 0; c < 4; ++c) qf[c] = frag_a_load(qptr + c * 32);

    v8f o[8] = {};
    float rmax[8], rsum[8];
    for (int r = 0; r < 8; ++r) { rmax[r] = -3.0e38f; rsum[r] = 0.f; }

    const bf16_t* kbase = Kh + (size_t)kvh * SEQL * HD;
    const bf16_t* vbase = Vt + (size_t)kvh * HD * SEQL;
    bf16_t* pw = &Ps[wave][0];

    for (int kv = 0; kv < qbase + 16; kv += 32) {
        if (kv + 32 < qbase + 16) {     // prefetch next KV tile (global_prefetch_b8)
            __builtin_prefetch(kbase + (size_t)(kv + 32 + nn) * HD, 0, 1);
            __builtin_prefetch(vbase + (size_t)nn * SEQL + kv + 32, 0, 1);
        }
        // S[16q x 32k] = Q * K^T : two 16x16 blocks, 4 WMMA each
        v8f sfr[2] = {};
        for (int j = 0; j < 2; ++j) {
            const bf16_t* kp = kbase + (size_t)(kv + j * 16 + nn) * HD;
            for (int c = 0; c < 4; ++c)
                sfr[j] = wmma_bf16(qf[c], frag_b_load(kp + c * 32), sfr[j]);
        }
        // causal mask + online softmax (row = qbase + r + 8*hh, col = kv+16j+nn)
        float alpha[8];
        for (int r = 0; r < 8; ++r) {
            int qrow = qbase + r + 8 * hh;
            float s0 = sfr[0][r] * ATT_SCALE;
            float s1 = sfr[1][r] * ATT_SCALE;
            if (kv + nn > qrow)      s0 = -3.0e38f;
            if (kv + 16 + nn > qrow) s1 = -3.0e38f;
            float m = fmaxf(s0, s1);
            for (int off = 8; off >= 1; off >>= 1)
                m = fmaxf(m, __shfl_xor(m, off, 16));
            float nm = fmaxf(rmax[r], m);
            float al = __expf(rmax[r] - nm);
            float p0 = __expf(s0 - nm);
            float p1 = __expf(s1 - nm);
            float ps = p0 + p1;
            for (int off = 8; off >= 1; off >>= 1)
                ps += __shfl_xor(ps, off, 16);
            rsum[r] = rsum[r] * al + ps;
            rmax[r] = nm;
            alpha[r] = al;
            pw[(r + 8 * hh) * 40 + nn]      = f2bf(p0);
            pw[(r + 8 * hh) * 40 + 16 + nn] = f2bf(p1);
        }
        for (int c = 0; c < 8; ++c)
            for (int r = 0; r < 8; ++r)
                o[c][r] *= alpha[r];
        // O += P * V : P re-read from LDS in A-layout, V B-frags key-contiguous
        v16bf pf = frag_a_load(pw + nn * 40);
        for (int c = 0; c < 8; ++c) {
            const bf16_t* vp = vbase + (size_t)(c * 16 + nn) * SEQL + kv;
            o[c] = wmma_bf16(pf, frag_b_load(vp), o[c]);
        }
    }

    for (int r = 0; r < 8; ++r) rsum[r] = 1.f / rsum[r];
    for (int c = 0; c < 8; ++c)
        for (int r = 0; r < 8; ++r) {
            int row = qbase + r + 8 * hh;
            Out[(size_t)row * (NH * HD) + head * HD + c * 16 + nn] =
                f2bf(o[c][r] * rsum[r]);
        }
}

// ---------------------------------------------------------------------------
// Host launcher
// ---------------------------------------------------------------------------
extern "C" void kernel_launch(void* const* d_in, const int* in_sizes, int n_in,
                              void* d_out, int out_size, void* d_ws, size_t ws_size,
                              hipStream_t stream) {
    (void)in_sizes; (void)n_in; (void)out_size; (void)ws_size;
    const float* x    = (const float*)d_in[0];
    const float* cosf = (const float*)d_in[1];
    const float* sinf = (const float*)d_in[2];
    // d_in[3] = positions (implicit), d_in[4] = mask (causal, computed analytically)
    const float* wq   = (const float*)d_in[5];
    const float* wk   = (const float*)d_in[6];
    const float* wv   = (const float*)d_in[7];
    const float* wo   = (const float*)d_in[8];
    float* out = (float*)d_out;

    // workspace carve-out
    char* p = (char*)d_ws;
    bf16_t* xb  = (bf16_t*)p; p += (size_t)SEQL * DIMM * 2;
    bf16_t* wqb = (bf16_t*)p; p += (size_t)DIMM * DIMM * 2;
    bf16_t* wkb = (bf16_t*)p; p += (size_t)(NKV * HD) * DIMM * 2;
    bf16_t* wvb = (bf16_t*)p; p += (size_t)(NKV * HD) * DIMM * 2;
    bf16_t* wob = (bf16_t*)p; p += (size_t)DIMM * DIMM * 2;
    float*  qf  = (float*)p;  p += (size_t)SEQL * DIMM * 4;
    float*  kf  = (float*)p;  p += (size_t)SEQL * (NKV * HD) * 4;
    float*  vf  = (float*)p;  p += (size_t)SEQL * (NKV * HD) * 4;
    bf16_t* qh  = (bf16_t*)p; p += (size_t)NH * SEQL * HD * 2;
    bf16_t* kh  = (bf16_t*)p; p += (size_t)NKV * SEQL * HD * 2;
    bf16_t* vth = (bf16_t*)p; p += (size_t)NKV * HD * SEQL * 2;
    bf16_t* ab  = (bf16_t*)p; p += (size_t)SEQL * DIMM * 2;

    // 1) casts to bf16
    cast_bf16_kernel<<<(SEQL * DIMM) / 1024, 256, 0, stream>>>(x, xb, SEQL * DIMM);
    cast_bf16_kernel<<<(DIMM * DIMM) / 1024, 256, 0, stream>>>(wq, wqb, DIMM * DIMM);
    cast_bf16_kernel<<<(NKV * HD * DIMM) / 1024, 256, 0, stream>>>(wk, wkb, NKV * HD * DIMM);
    cast_bf16_kernel<<<(NKV * HD * DIMM) / 1024, 256, 0, stream>>>(wv, wvb, NKV * HD * DIMM);
    cast_bf16_kernel<<<(DIMM * DIMM) / 1024, 256, 0, stream>>>(wo, wob, DIMM * DIMM);

    // 2) QKV projections (C = x * W^T)
    gemm_bf16_kernel<<<dim3(DIMM / GBN, SEQL / GBM), 256, 0, stream>>>(
        xb, wqb, qf, SEQL, DIMM, DIMM);
    gemm_bf16_kernel<<<dim3((NKV * HD) / GBN, SEQL / GBM), 256, 0, stream>>>(
        xb, wkb, kf, SEQL, NKV * HD, DIMM);
    gemm_bf16_kernel<<<dim3((NKV * HD) / GBN, SEQL / GBM), 256, 0, stream>>>(
        xb, wvb, vf, SEQL, NKV * HD, DIMM);

    // 3) RoPE + head-major repack, V transpose
    rope_pack_kernel<<<(SEQL * NH * 64) / 256, 256, 0, stream>>>(qf, cosf, sinf, qh, NH);
    rope_pack_kernel<<<(SEQL * NKV * 64) / 256, 256, 0, stream>>>(kf, cosf, sinf, kh, NKV);
    pack_vt_kernel<<<(SEQL * NKV * HD) / 256, 256, 0, stream>>>(vf, vth);

    // 4) flash attention: 32 heads x 128 q-tiles (4 waves / block)
    attn_kernel<<<dim3(NH, SEQL / 16 / 4), 128, 0, stream>>>(qh, kh, vth, ab);

    // 5) output projection
    gemm_bf16_kernel<<<dim3(DIMM / GBN, SEQL / GBM), 256, 0, stream>>>(
        ab, wob, out, SEQL, DIMM, DIMM);
}
